// Token_dend_QK_Attention_12953621364892
// MI455X (gfx1250) — matrix operand order repaired and checked by reading, not compiled, not run on hardware
//
#include <hip/hip_runtime.h>
#include <hip/hip_bf16.h>

typedef float    v2f  __attribute__((ext_vector_type(2)));
typedef float    v8f  __attribute__((ext_vector_type(8)));
typedef _Float16 v16h __attribute__((ext_vector_type(16)));

#define TSTEPS 4
#define BATCH  16
#define CH     512
#define HWPIX  1024
#define NPIX   (BATCH * HWPIX)   // 16384 pixels per timestep

// workspace layout (float offsets)
#define OFF_QWT  0                         // q weights, BN-scaled, [K][N] f32
#define OFF_KWT  (512 * 512)               // k weights
#define OFF_QB   (2 * 512 * 512)           // folded biases
#define OFF_KB   (2 * 512 * 512 + 512)
#define OFF_PB   (2 * 512 * 512 + 1024)
#define OFF_PWHI (2 * 512 * 512 + 1536)    // proj w hi, f16 B-fragment layout
#define OFF_PWLO (OFF_PWHI + 512 * 512 / 2)// proj w lo
#define OFF_Y    (1 << 20)                 // gated spikes, f16 A-fragment layout

// ---------------------------------------------------------------------------
// Prep (q/k): fold BN scale into transposed f32 weights + folded bias.
// wT[cin][cout] = w[cout][cin] * gamma/sqrt(var+eps)
// ---------------------------------------------------------------------------
__global__ __launch_bounds__(256)
void prep_kernel(const float* __restrict__ w,
                 const float* __restrict__ gamma,
                 const float* __restrict__ beta,
                 const float* __restrict__ mean,
                 const float* __restrict__ var,
                 float* __restrict__ wT,
                 float* __restrict__ bias)
{
    const int idx  = blockIdx.x * 256 + threadIdx.x;   // 0 .. CH*CH-1
    const int cout = idx & (CH - 1);
    const int cin  = idx >> 9;
    const float inv = gamma[cout] * rsqrtf(var[cout] + 1e-5f);
    wT[(size_t)cin * CH + cout] = w[(size_t)cout * CH + cin] * inv;
    if (cin == 0) bias[cout] = beta[cout] - mean[cout] * inv;
}

// ---------------------------------------------------------------------------
// Prep (proj): BN+bias fold, then split into hi/lo f16 stored directly in the
// WMMA f16 B-fragment layout: group g = ks*32 + (n/16); within a group,
// lane = (n&15) + 16*(K-half); element s = K & 15. (contiguous k, same order as the A pack: lanes 0-15 hold
// K=0..15, lanes 16-31 hold K=16..31, N = lane&15.)
// ---------------------------------------------------------------------------
__global__ __launch_bounds__(256)
void prep_proj(const float* __restrict__ w,
               const float* __restrict__ gamma,
               const float* __restrict__ beta,
               const float* __restrict__ mean,
               const float* __restrict__ var,
               const float* __restrict__ convb,
               float* __restrict__ bias,
               _Float16* __restrict__ bhi,
               _Float16* __restrict__ blo)
{
    const int idx = blockIdx.x * 256 + threadIdx.x;    // 0 .. CH*CH-1
    const int n   = idx & (CH - 1);                    // cout (N)
    const int K   = idx >> 9;                          // cin  (K)
    const float inv = gamma[n] * rsqrtf(var[n] + 1e-5f);
    const float v   = w[(size_t)n * CH + K] * inv;
    const _Float16 hi = (_Float16)v;
    const _Float16 lo = (_Float16)(v - (float)hi);
    const int ks = K >> 5, kr = K & 31, h = kr >> 4, s = kr & 15;
    const int ntile = n >> 4, l = n & 15, lanef = l + 16 * h;
    const size_t off = (((size_t)(ks * 32 + ntile)) * 32 + lanef) * 16 + s;
    bhi[off] = hi;
    blo[off] = lo;
    if (K == 0) bias[n] = (convb[n] - mean[n]) * inv + beta[n];
}

// ---------------------------------------------------------------------------
// Kernel 1: fused Q/K conv(BN-folded, f32 WMMA 16x16x4) -> LIF -> head-sum ->
// attn LIF -> gate. Block: 64 pixels x 128 channels (2 heads), 8 waves; wave
// (mt, half) owns 16 pixels x one whole 64-ch head for BOTH branches (shared
// A fragments). LIF states live in VGPRs across the in-kernel t loop.
// Output: gated k-spikes packed as f16 WMMA A-fragments (lane-contiguous 32B).
// ---------------------------------------------------------------------------
__global__ __launch_bounds__(256, 1)
void qk_attn_kernel(const float* __restrict__ x,     // [T][B][CH][HW]
                    const float* __restrict__ qwT,   // [CH][CH] BN-scaled
                    const float* __restrict__ kwT,
                    const float* __restrict__ qb,    // [CH] folded bias
                    const float* __restrict__ kb,
                    v16h* __restrict__ yfrag)        // f16 A-fragment layout
{
    __shared__ float lds[128 * 65];

    const int tid  = threadIdx.x;
    const int wave = tid >> 5;
    const int lane = tid & 31;
    const int l15  = lane & 15;
    const int h16  = lane >> 4;
    const int mt   = wave & 3;
    const int half = wave >> 2;

    const int cblk  = blockIdx.x * 128;        // channel-major grid: blocks
    const int pblk  = blockIdx.y * 64;         // sharing x are launch-adjacent
    const int b     = pblk >> 10;
    const int hw0   = pblk & (HWPIX - 1);
    const int mOff  = hw0 + mt * 16 + l15;
    const int nbase = cblk + half * 64;        // wave's 64 channels (1 head)

    float biasq[4], biask[4];
#pragma unroll
    for (int nt = 0; nt < 4; ++nt) {
        biasq[nt] = qb[nbase + nt * 16 + l15];
        biask[nt] = kb[nbase + nt * 16 + l15];
    }

    const v8f zero8 = {0.f, 0.f, 0.f, 0.f, 0.f, 0.f, 0.f, 0.f};
    v8f vq[4], vk[4];
#pragma unroll
    for (int nt = 0; nt < 4; ++nt) { vq[nt] = zero8; vk[nt] = zero8; }
    float v_attn[8];
#pragma unroll
    for (int v = 0; v < 8; ++v) v_attn[v] = 0.f;

    for (int t = 0; t < TSTEPS; ++t) {
        const float* xA = x + ((size_t)(t * BATCH + b)) * CH * HWPIX;
        v8f accq[4], acck[4];
#pragma unroll
        for (int nt = 0; nt < 4; ++nt) { accq[nt] = zero8; acck[nt] = zero8; }

#pragma unroll 2
        for (int k = 0; k < CH; k += 4) {
            const int kk = k + h16 * 2;
            v2f a;                                  // A 16x4: M=l15, K=kk..kk+1
            a.x = xA[(size_t)kk * HWPIX + mOff];
            a.y = xA[(size_t)(kk + 1) * HWPIX + mOff];
#pragma unroll
            for (int nt = 0; nt < 4; ++nt) {
                const int ncol = nbase + nt * 16 + l15;
                v2f bq, bk;                          // B 4x16: N=l15
                bq.x = qwT[(size_t)kk * CH + ncol];
                bq.y = qwT[(size_t)(kk + 1) * CH + ncol];
                bk.x = kwT[(size_t)kk * CH + ncol];
                bk.y = kwT[(size_t)(kk + 1) * CH + ncol];
                accq[nt] = __builtin_amdgcn_wmma_f32_16x16x4_f32(
                    false, a, false, bq, (short)0, accq[nt], false, false);
                acck[nt] = __builtin_amdgcn_wmma_f32_16x16x4_f32(
                    false, a, false, bk, (short)0, acck[nt], false, false);
            }
        }

        // --- bias + LIF (tau=2: v' = 0.5*(v + x)), hard reset ---
        float sumrow[8];
#pragma unroll
        for (int v = 0; v < 8; ++v) sumrow[v] = 0.f;
#pragma unroll
        for (int nt = 0; nt < 4; ++nt) {
#pragma unroll
            for (int v = 0; v < 8; ++v) {
                float vvq = 0.5f * (vq[nt][v] + accq[nt][v] + biasq[nt]);
                float spq = (vvq >= 1.0f) ? 1.0f : 0.0f;
                vq[nt][v] = vvq * (1.0f - spq);
                sumrow[v] += spq;

                float vvk = 0.5f * (vk[nt][v] + acck[nt][v] + biask[nt]);
                float spk = (vvk >= 1.0f) ? 1.0f : 0.0f;
                vk[nt][v] = vvk * (1.0f - spk);
                acck[nt][v] = spk;
            }
        }
        // head-sum of q spikes (wave owns the whole head) + attn LIF @0.5
        float attn_sp[8];
#pragma unroll
        for (int v = 0; v < 8; ++v) {
            float s = sumrow[v];
            s += __shfl_xor(s, 1, 32);
            s += __shfl_xor(s, 2, 32);
            s += __shfl_xor(s, 4, 32);
            s += __shfl_xor(s, 8, 32);
            float va = 0.5f * (v_attn[v] + s);
            float sa = (va >= 0.5f) ? 1.0f : 0.0f;
            v_attn[v] = va * (1.0f - sa);
            attn_sp[v] = sa;
        }
        // gate k spikes into the LDS transpose tile [c_local][p_local]
#pragma unroll
        for (int nt = 0; nt < 4; ++nt) {
#pragma unroll
            for (int v = 0; v < 8; ++v) {
                const int cl = half * 64 + nt * 16 + l15;
                const int pl = mt * 16 + v + 8 * h16;
                lds[cl * 65 + pl] = attn_sp[v] * acck[nt][v];
            }
        }
        __syncthreads();
        // pack into f16 WMMA A-fragments: 16 frag tiles (4 ks x 4 pt) x 32
        // lanes = 512 entries; each thread packs 2 lane-entries (32B each).
#pragma unroll
        for (int e = 0; e < 2; ++e) {
            const int ent   = tid * 2 + e;
            const int lanef = ent & 31;
            const int ft    = ent >> 5;        // 0..15
            const int ksl   = ft >> 2;         // local 32-channel group
            const int ptl   = ft & 3;          // local 16-pixel group
            const int m     = lanef & 15;
            const int hh    = lanef >> 4;
            v16h pack;
#pragma unroll
            for (int s = 0; s < 16; ++s) {
                // contiguous k order, same as B pack: half 0 -> K in {0..15},
                //                     half 1 -> K in {16..31}
                const int kpos = s + 16 * hh;
                pack[s] = (_Float16)lds[(ksl * 32 + kpos) * 65 + ptl * 16 + m];
            }
            const size_t g = ((size_t)t * 16 + (cblk >> 5) + ksl) * 1024
                             + (pblk >> 4) + ptl;
            yfrag[g * 32 + lanef] = pack;
        }
        __syncthreads();
    }
}

// ---------------------------------------------------------------------------
// Kernel 2: proj conv via f16 WMMA 16x16x32 with hi/lo split weights
// (A = binary spikes, exact in f16 => result matches f32 to ~2^-22).
// t-loop is INSIDE the K-loop: B fragments loaded once, reused for all 4 t.
// Block: 64 pixels x 64 channels, 8 waves; wave = 16 pixels x 32 channels.
// ---------------------------------------------------------------------------
__global__ __launch_bounds__(256, 1)
void proj_kernel(const v16h* __restrict__ yfrag,
                 const v16h* __restrict__ bhi,
                 const v16h* __restrict__ blo,
                 const float* __restrict__ pb,
                 float* __restrict__ out)          // [T][B][CH][HW]
{
    __shared__ float lds[64 * 65];

    const int tid  = threadIdx.x;
    const int wave = tid >> 5;
    const int lane = tid & 31;
    const int l15  = lane & 15;
    const int h16  = lane >> 4;
    const int mt   = wave & 3;
    const int half = wave >> 2;

    const int cblk = blockIdx.x * 64;
    const int pblk = blockIdx.y * 64;
    const int b    = pblk >> 10;
    const int hw0  = pblk & (HWPIX - 1);
    const int ptg  = (pblk >> 4) + mt;            // global 16-pixel tile
    const int nbase = cblk + half * 32;
    const int ngrp  = nbase >> 4;                 // global 16-channel tile

    float bias[2];
#pragma unroll
    for (int nt = 0; nt < 2; ++nt) bias[nt] = pb[nbase + nt * 16 + l15];

    const v8f zero8 = {0.f, 0.f, 0.f, 0.f, 0.f, 0.f, 0.f, 0.f};
    v8f vs[2];
    vs[0] = zero8; vs[1] = zero8;
    v8f acc[2][TSTEPS];
#pragma unroll
    for (int nt = 0; nt < 2; ++nt)
#pragma unroll
        for (int t = 0; t < TSTEPS; ++t) acc[nt][t] = zero8;

    for (int ks = 0; ks < 16; ++ks) {             // K = 512 in steps of 32
        v16h a[TSTEPS];
#pragma unroll
        for (int t = 0; t < TSTEPS; ++t)
            a[t] = yfrag[(((size_t)t * 16 + ks) * 1024 + ptg) * 32 + lane];
#pragma unroll
        for (int nt = 0; nt < 2; ++nt) {
            const size_t bi = ((size_t)(ks * 32 + ngrp + nt)) * 32 + lane;
            const v16h bh = bhi[bi];
            const v16h bl = blo[bi];
#pragma unroll
            for (int t = 0; t < TSTEPS; ++t) {
                acc[nt][t] = __builtin_amdgcn_wmma_f32_16x16x32_f16(
                    false, a[t], false, bh, (short)0, acc[nt][t], false, false);
                acc[nt][t] = __builtin_amdgcn_wmma_f32_16x16x32_f16(
                    false, a[t], false, bl, (short)0, acc[nt][t], false, false);
            }
        }
    }

    // sequential LIF over t, LDS transpose, coalesced store
    for (int t = 0; t < TSTEPS; ++t) {
#pragma unroll
        for (int nt = 0; nt < 2; ++nt) {
#pragma unroll
            for (int v = 0; v < 8; ++v) {
                float vv = 0.5f * (vs[nt][v] + acc[nt][t][v] + bias[nt]);
                float sp = (vv >= 1.0f) ? 1.0f : 0.0f;
                vs[nt][v] = vv * (1.0f - sp);
                const int cl = half * 32 + nt * 16 + l15;
                const int pl = mt * 16 + v + 8 * h16;
                lds[cl * 65 + pl] = sp;
            }
        }
        __syncthreads();
#pragma unroll 4
        for (int it = 0; it < 16; ++it) {
            const int cl = it * 4 + (tid >> 6);
            const int pl = tid & 63;
            out[((size_t)(t * BATCH + b) * CH + cblk + cl) * HWPIX + hw0 + pl] =
                lds[cl * 65 + pl];
        }
        __syncthreads();
    }
}

// ---------------------------------------------------------------------------
extern "C" void kernel_launch(void* const* d_in, const int* in_sizes, int n_in,
                              void* d_out, int out_size, void* d_ws, size_t ws_size,
                              hipStream_t stream)
{
    const float* x    = (const float*)d_in[0];
    const float* q_w  = (const float*)d_in[1];
    const float* q_g  = (const float*)d_in[2];
    const float* q_b  = (const float*)d_in[3];
    const float* q_m  = (const float*)d_in[4];
    const float* q_v  = (const float*)d_in[5];
    const float* k_w  = (const float*)d_in[6];
    const float* k_g  = (const float*)d_in[7];
    const float* k_b  = (const float*)d_in[8];
    const float* k_m  = (const float*)d_in[9];
    const float* k_v  = (const float*)d_in[10];
    const float* p_w  = (const float*)d_in[11];
    const float* p_cb = (const float*)d_in[12];
    const float* p_g  = (const float*)d_in[13];
    const float* p_b  = (const float*)d_in[14];
    const float* p_m  = (const float*)d_in[15];
    const float* p_v  = (const float*)d_in[16];

    float* ws      = (float*)d_ws;
    float* qwT     = ws + OFF_QWT;
    float* kwT     = ws + OFF_KWT;
    float* qbias   = ws + OFF_QB;
    float* kbias   = ws + OFF_KB;
    float* pbias   = ws + OFF_PB;
    _Float16* pwhi = (_Float16*)(ws + OFF_PWHI);
    _Float16* pwlo = (_Float16*)(ws + OFF_PWLO);
    v16h* yfrag    = (v16h*)(ws + OFF_Y);

    const dim3 prepGrid(CH * CH / 256);
    prep_kernel<<<prepGrid, 256, 0, stream>>>(q_w, q_g, q_b, q_m, q_v, qwT, qbias);
    prep_kernel<<<prepGrid, 256, 0, stream>>>(k_w, k_g, k_b, k_m, k_v, kwT, kbias);
    prep_proj  <<<prepGrid, 256, 0, stream>>>(p_w, p_g, p_b, p_m, p_v, p_cb,
                                              pbias, pwhi, pwlo);

    const dim3 grid1(CH / 128, NPIX / 64);   // channel-major for L2 reuse of x
    qk_attn_kernel<<<grid1, 256, 0, stream>>>(x, qwT, kwT, qbias, kbias, yfrag);

    const dim3 grid2(CH / 64, NPIX / 64);
    proj_kernel<<<grid2, 256, 0, stream>>>(yfrag,
                                           (const v16h*)pwhi,
                                           (const v16h*)pwlo,
                                           pbias,
                                           (float*)d_out);
}
